// ResMoELoRALinear_71150428225587
// MI455X (gfx1250) — matrix operand order repaired and checked by reading, not compiled, not run on hardware
//
#include <hip/hip_runtime.h>
#include <hip/hip_bf16.h>

// ---------------------------------------------------------------------------
// ResMoE-LoRA fused linear for gfx1250 (MI455X).
// out = x @ base_w^T + bias + SCALING * (topk-combine) . (x@A^T) . B
// One bf16 WMMA GEMM with K = 4096 + 16*64 = 5120:
//   A' = [ bf16(x) | C ]   where C[m, e*64+r] = w_k(m) * h[m,r] (top-2 sparse)
//   B' = [ bf16(base_w) | bf16(SCALING * B_flat) ]
// GEMM uses double-buffered LDS fed by global_load_async_to_lds_b128 (ASYNCcnt).
// ---------------------------------------------------------------------------

typedef __bf16 bf16_t;
typedef __attribute__((ext_vector_type(16))) __bf16 v16bf;
typedef __attribute__((ext_vector_type(8)))  __bf16 v8bf;
typedef __attribute__((ext_vector_type(8)))  float  v8f;

#define DIN   4096
#define DOUT  4096
#define RANK  64
#define NEXP  16
#define KTOT  (DIN + NEXP * RANK)   // 5120
#define MTOK  8192                  // 4 * 2048 tokens
#define SCALING 0.5f

#define BM 128
#define BN 128
#define BK 32
#define LDK (BK + 8)                  // padded LDS row stride (bf16), 80B: 16B aligned
#define STAGE_ELEMS ((BM + BN) * LDK) // bf16 elems per stage
#define STAGE_BYTES (STAGE_ELEMS * 2) // 20480 B per stage, 2 stages = 40 KB
#define NK (KTOT / BK)                // 160 K-steps

// ---------------- Kernel 1: fused bf16 weight pack [DOUT][KTOT] -------------
__global__ __launch_bounds__(256)
void pack_weights(const float* __restrict__ base_w,
                  const float* __restrict__ Bmat,     // [E][DOUT][RANK]
                  bf16_t* __restrict__ wb) {
  long id = (long)blockIdx.x * 256 + threadIdx.x;
  const long total = (long)DOUT * KTOT;
  if (id >= total) return;
  const int n = (int)(id / KTOT);
  const int k = (int)(id % KTOT);
  float v;
  if (k < DIN) {
    v = base_w[(long)n * DIN + k];
  } else {
    const int j = k - DIN;
    const int e = j >> 6;
    const int r = j & 63;
    v = SCALING * Bmat[((long)e * DOUT + n) * RANK + r];
  }
  wb[id] = (bf16_t)v;
}

// ------- Kernel 2: per-token router + reservoir + bf16 A' row pack ----------
__global__ __launch_bounds__(256)
void route_pack_tokens(const float* __restrict__ x,
                       const float* __restrict__ Ares,     // [RANK][DIN]
                       const float* __restrict__ router_w, // [NEXP][DIN]
                       bf16_t* __restrict__ xc) {          // [MTOK][KTOT]
  __shared__ float xs[DIN];
  __shared__ float out80[NEXP + RANK];
  __shared__ float sw[2];
  __shared__ int   se[2];

  const int m    = blockIdx.x;
  const int tid  = threadIdx.x;
  const int lane = tid & 31;
  const int wave = tid >> 5;

  const float* xr  = x  + (long)m * DIN;
  bf16_t*      xcr = xc + (long)m * KTOT;

  // Stage x row into LDS and emit bf16 copy into A'[m, 0:DIN].
  for (int i = tid; i < DIN / 8; i += 256) {
    const float4 f0 = ((const float4*)xr)[2 * i];
    const float4 f1 = ((const float4*)xr)[2 * i + 1];
    xs[8*i+0]=f0.x; xs[8*i+1]=f0.y; xs[8*i+2]=f0.z; xs[8*i+3]=f0.w;
    xs[8*i+4]=f1.x; xs[8*i+5]=f1.y; xs[8*i+6]=f1.z; xs[8*i+7]=f1.w;
    union { bf16_t h[8]; uint4 u; } t;
    t.h[0]=(bf16_t)f0.x; t.h[1]=(bf16_t)f0.y; t.h[2]=(bf16_t)f0.z; t.h[3]=(bf16_t)f0.w;
    t.h[4]=(bf16_t)f1.x; t.h[5]=(bf16_t)f1.y; t.h[6]=(bf16_t)f1.z; t.h[7]=(bf16_t)f1.w;
    ((uint4*)xcr)[i] = t.u;
  }
  // Zero the MoE coefficient tail A'[m, DIN:KTOT].
  for (int i = tid; i < (NEXP * RANK) / 8; i += 256) {
    ((uint4*)(xcr + DIN))[i] = make_uint4(0u, 0u, 0u, 0u);
  }
  __syncthreads();

  // 80 dot products (16 router logits + 64 reservoir h), one per wave-round.
  for (int o = wave; o < NEXP + RANK; o += 8) {
    const float* wrow = (o < NEXP) ? (router_w + (long)o * DIN)
                                   : (Ares + (long)(o - NEXP) * DIN);
    float s = 0.0f;
    for (int k = lane; k < DIN; k += 32) s = fmaf(xs[k], wrow[k], s);
    #pragma unroll
    for (int off = 16; off; off >>= 1) s += __shfl_xor(s, off, 32);
    if (lane == 0) out80[o] = s;
  }
  __syncthreads();

  // Softmax + top-2 (tiny: one thread).
  if (tid == 0) {
    float mx = out80[0];
    for (int e = 1; e < NEXP; ++e) mx = fmaxf(mx, out80[e]);
    float p[NEXP]; float sum = 0.0f;
    for (int e = 0; e < NEXP; ++e) { p[e] = __expf(out80[e] - mx); sum += p[e]; }
    int i0 = 0;
    for (int e = 1; e < NEXP; ++e) if (p[e] > p[i0]) i0 = e;
    int i1 = (i0 == 0) ? 1 : 0;
    for (int e = 0; e < NEXP; ++e) if (e != i0 && p[e] > p[i1]) i1 = e;
    const float w0 = p[i0] / sum, w1 = p[i1] / sum;
    const float nrm = 1.0f / (w0 + w1 + 1e-6f);
    sw[0] = w0 * nrm; sw[1] = w1 * nrm;
    se[0] = i0;       se[1] = i1;
  }
  __syncthreads();

  // Scatter C coefficients for the two selected experts.
  if (tid < 2 * RANK) {
    const int r     = tid & (RANK - 1);
    const int which = tid >> 6;
    const int e     = se[which];
    const float w   = sw[which];
    xcr[DIN + e * RANK + r] = (bf16_t)(w * out80[NEXP + r]);
  }
}

// ---------------- Kernel 3: bf16 WMMA GEMM out = A' @ B'^T + bias -----------
// Double-buffered LDS, async global->LDS staging (ASYNCcnt pipeline).
__global__ __launch_bounds__(256)
void gemm_wmma(const bf16_t* __restrict__ Amat,   // [MTOK][KTOT]
               const bf16_t* __restrict__ Bmat,   // [DOUT][KTOT]
               const float*  __restrict__ bias,   // [DOUT]
               float* __restrict__ out) {         // [MTOK][DOUT]
  __shared__ bf16_t lds[2 * STAGE_ELEMS];

  const int tid  = threadIdx.x;
  const int lane = tid & 31;
  const int wave = tid >> 5;
  const int half = lane >> 4;      // lane 16..31 -> 1
  const int l16  = lane & 15;

  const int m0 = blockIdx.y * BM;
  const int n0 = blockIdx.x * BN;
  const int wm = (wave >> 2) * 64; // 2 waves along M
  const int wn = (wave & 3) * 32;  // 4 waves along N

  // --- per-thread async-copy descriptors (4 x 16B chunks per stage) --------
  // A tile: 512 chunks (row = c>>2, col8 = (c&3)*8); threads own c = tid, tid+256.
  const int cA0 = tid, cA1 = tid + 256;
  const unsigned lds_u32 = (unsigned)(size_t)(&lds[0]);   // LDS offset = low 32 bits of the flat address
  const unsigned ldsA0 = lds_u32 + (unsigned)(((cA0 >> 2) * LDK + (cA0 & 3) * 8) * 2);
  const unsigned ldsA1 = lds_u32 + (unsigned)(((cA1 >> 2) * LDK + (cA1 & 3) * 8) * 2);
  const unsigned ldsB0 = lds_u32 + (unsigned)((BM * LDK + (cA0 >> 2) * LDK + (cA0 & 3) * 8) * 2);
  const unsigned ldsB1 = lds_u32 + (unsigned)((BM * LDK + (cA1 >> 2) * LDK + (cA1 & 3) * 8) * 2);
  const unsigned long long gA = (unsigned long long)Amat;
  const unsigned long long gB = (unsigned long long)Bmat;
  unsigned voffA0 = (unsigned)(((long)(m0 + (cA0 >> 2)) * KTOT + (cA0 & 3) * 8) * 2);
  unsigned voffA1 = (unsigned)(((long)(m0 + (cA1 >> 2)) * KTOT + (cA1 & 3) * 8) * 2);
  unsigned voffB0 = (unsigned)(((long)(n0 + (cA0 >> 2)) * KTOT + (cA0 & 3) * 8) * 2);
  unsigned voffB1 = (unsigned)(((long)(n0 + (cA1 >> 2)) * KTOT + (cA1 & 3) * 8) * 2);

  #define ISSUE_STAGE(BUFOFF)                                                   \
    do {                                                                        \
      asm volatile("global_load_async_to_lds_b128 %0, %1, %2"                   \
                   :: "v"(ldsA0 + (BUFOFF)), "v"(voffA0), "s"(gA) : "memory");  \
      asm volatile("global_load_async_to_lds_b128 %0, %1, %2"                   \
                   :: "v"(ldsA1 + (BUFOFF)), "v"(voffA1), "s"(gA) : "memory");  \
      asm volatile("global_load_async_to_lds_b128 %0, %1, %2"                   \
                   :: "v"(ldsB0 + (BUFOFF)), "v"(voffB0), "s"(gB) : "memory");  \
      asm volatile("global_load_async_to_lds_b128 %0, %1, %2"                   \
                   :: "v"(ldsB1 + (BUFOFF)), "v"(voffB1), "s"(gB) : "memory");  \
      voffA0 += BK * 2; voffA1 += BK * 2; voffB0 += BK * 2; voffB1 += BK * 2;   \
    } while (0)

  v8f acc[4][2] = {};              // 64x32 wave tile = 4x2 subtiles of 16x16

  ISSUE_STAGE(0u);                 // prologue: stage 0 in flight

  for (int it = 0; it < NK; ++it) {
    const int buf = it & 1;
    if (it + 1 < NK) {
      // WAR on buf^1 is safe: previous iteration's trailing barrier ensured
      // every wave finished its ds reads of buf^1 before we got here.
      ISSUE_STAGE(buf ? 0u : (unsigned)STAGE_BYTES);
      asm volatile("s_wait_asynccnt 4" ::: "memory");   // current stage landed
    } else {
      asm volatile("s_wait_asynccnt 0" ::: "memory");
    }
    __syncthreads();               // all waves' stage-it loads visible

    const bf16_t* bufp  = lds + buf * STAGE_ELEMS;
    const bf16_t* bufpB = bufp + BM * LDK;

    // A fragment (16-bit A 16x32 layout): lane holds row M=l16;
    // elems 0..7 = K[8*half..+7], elems 8..15 = K[16+8*half..+7].
    v16bf afrag[4];
    #pragma unroll
    for (int mi = 0; mi < 4; ++mi) {
      const bf16_t* p = bufp + (wm + mi * 16 + l16) * LDK + half * 8;
      v8bf lo = *(const v8bf*)(p);
      v8bf hi = *(const v8bf*)(p + 16);
      afrag[mi] = __builtin_shufflevector(lo, hi, 0,1,2,3,4,5,6,7,
                                                  8,9,10,11,12,13,14,15);
    }
    // B fragment (16-bit B 32x16 layout): lane holds col N=l16;
    // elems 0..7 = K[8*half..+7], elems 8..15 = K[16+8*half..+7].
    v16bf bfrag[2];
    #pragma unroll
    for (int ni = 0; ni < 2; ++ni) {
      const bf16_t* p = bufpB + (wn + ni * 16 + l16) * LDK + half * 8;
      v8bf lo = *(const v8bf*)(p);
      v8bf hi = *(const v8bf*)(p + 16);
      bfrag[ni] = __builtin_shufflevector(lo, hi, 0,1,2,3,4,5,6,7,
                                                  8,9,10,11,12,13,14,15);
    }

    #pragma unroll
    for (int mi = 0; mi < 4; ++mi)
      #pragma unroll
      for (int ni = 0; ni < 2; ++ni)
        acc[mi][ni] = __builtin_amdgcn_wmma_f32_16x16x32_bf16(
            false, afrag[mi], false, bfrag[ni],
            (short)0, acc[mi][ni], false, false);

    __syncthreads();               // WAR fence before next issue overwrites buf^1
  }
  #undef ISSUE_STAGE

  // Store: C/D layout -> row m = g + 8*half, col n = l16. Streamed once: NT.
  #pragma unroll
  for (int ni = 0; ni < 2; ++ni) {
    const int col = n0 + wn + ni * 16 + l16;
    const float b = bias[col];
    #pragma unroll
    for (int mi = 0; mi < 4; ++mi) {
      #pragma unroll
      for (int g = 0; g < 8; ++g) {
        const int row = m0 + wm + mi * 16 + g + half * 8;
        __builtin_nontemporal_store(acc[mi][ni][g] + b, out + (long)row * DOUT + col);
      }
    }
  }
}

// ---------------------------------------------------------------------------
extern "C" void kernel_launch(void* const* d_in, const int* in_sizes, int n_in,
                              void* d_out, int out_size, void* d_ws, size_t ws_size,
                              hipStream_t stream) {
  const float* x        = (const float*)d_in[0];
  const float* base_w   = (const float*)d_in[1];
  const float* base_b   = (const float*)d_in[2];
  const float* Ares     = (const float*)d_in[3];
  const float* Bmat     = (const float*)d_in[4];
  const float* router_w = (const float*)d_in[5];
  float* out = (float*)d_out;

  // Workspace: A' bf16 [MTOK][KTOT] then B' bf16 [DOUT][KTOT] (~120 MB total).
  bf16_t* xc = (bf16_t*)d_ws;
  bf16_t* wb = (bf16_t*)((char*)d_ws + (size_t)MTOK * KTOT * sizeof(bf16_t));

  {
    const long total = (long)DOUT * KTOT;
    const int blocks = (int)((total + 255) / 256);
    pack_weights<<<blocks, 256, 0, stream>>>(base_w, Bmat, wb);
  }
  route_pack_tokens<<<MTOK, 256, 0, stream>>>(x, Ares, router_w, xc);

  dim3 grid(DOUT / BN, MTOK / BM);
  gemm_wmma<<<grid, 256, 0, stream>>>(xc, wb, base_b, out);
}
